// Block_55052890800142
// MI455X (gfx1250) — matrix operand order repaired and checked by reading, not compiled, not run on hardware
//
#include <hip/hip_runtime.h>

typedef __attribute__((ext_vector_type(16))) _Float16 v16h;
typedef __attribute__((ext_vector_type(8)))  _Float16 v8h;
typedef __attribute__((ext_vector_type(8)))  float    v8f;
typedef __attribute__((ext_vector_type(4)))  int      v4i;

#define EMB   1024
#define HEADS 16
#define HD    64
#define FFDIM 2048
#define SEQ   2048
#define BATCH 4
#define ROWS  (BATCH*SEQ)   /* 8192 */

#if __has_builtin(__builtin_amdgcn_global_load_async_to_lds_b128)
#define ASYNC_OK 1
#else
#define ASYNC_OK 0
#endif

#if ASYNC_OK
static __device__ __forceinline__ void async_b128(const void* g, void* l) {
    __builtin_amdgcn_global_load_async_to_lds_b128(
        (__attribute__((address_space(1))) v4i*)g,
        (__attribute__((address_space(3))) v4i*)l, 0, 0);
}
#endif

static __device__ __forceinline__ void wait_async0() {
#if __has_builtin(__builtin_amdgcn_s_wait_asynccnt)
    __builtin_amdgcn_s_wait_asynccnt(0);
#else
    asm volatile("s_wait_asynccnt 0x0" ::: "memory");
#endif
}

// Build a v16h operand from two contiguous 16-byte LDS chunks (2x ds_load_b128)
static __device__ __forceinline__ v16h frag2(const _Float16* p0, const _Float16* p1) {
    v8h lo = *(const v8h*)p0;
    v8h hi = *(const v8h*)p1;
    v16h r;
    #pragma unroll
    for (int i = 0; i < 8; ++i) { r[i] = lo[i]; r[8 + i] = hi[i]; }
    return r;
}

// ---------------------------------------------------------------------------
// Weight conversion: f32 -> f16, transposed to [N][K] (n-major) for B operand
// ---------------------------------------------------------------------------
__global__ void __launch_bounds__(256)
cvt_t_kernel(const float* __restrict__ src, _Float16* __restrict__ dst, int N, int K) {
    int i = blockIdx.x * 256 + threadIdx.x;   // over N*K
    int n = i / K;
    int k = i - n * K;
    dst[(size_t)n * K + k] = (_Float16)src[(size_t)k * N + n];
}

// wq/wk/wv are [H][E][HD]; pack transposed into Wqkv_t [3*H*HD][E]
__global__ void __launch_bounds__(256)
cvt_qkv_t_kernel(const float* __restrict__ wq, const float* __restrict__ wk,
                 const float* __restrict__ wv, _Float16* __restrict__ W) {
    int i = blockIdx.x * 256 + threadIdx.x;   // over H*E*HD = 1M
    int h = i >> 16;          // / (EMB*HD)
    int r = i & 65535;
    int e = r >> 6;
    int d = r & 63;
    size_t o = (size_t)(h * HD + d) * EMB + e;
    W[o]                            = (_Float16)wq[i];
    W[o + (size_t)EMB * EMB]        = (_Float16)wk[i];
    W[o + (size_t)2 * EMB * EMB]    = (_Float16)wv[i];
}

// ---------------------------------------------------------------------------
// LayerNorm: rows of 1024, f32 in -> f16 out
// ---------------------------------------------------------------------------
__global__ void __launch_bounds__(256)
layernorm_kernel(const float* __restrict__ x, const float* __restrict__ g,
                 const float* __restrict__ b, _Float16* __restrict__ out) {
    __shared__ float red[256];
    const int row = blockIdx.x;
    const int tid = threadIdx.x;
    const float* xr = x + (size_t)row * EMB;
    float v[4]; float s = 0.f;
    #pragma unroll
    for (int i = 0; i < 4; ++i) { v[i] = xr[tid + 256 * i]; s += v[i]; }
    red[tid] = s; __syncthreads();
    for (int off = 128; off > 0; off >>= 1) {
        if (tid < off) red[tid] += red[tid + off];
        __syncthreads();
    }
    float mu = red[0] * (1.f / EMB);
    __syncthreads();
    float s2 = 0.f;
    #pragma unroll
    for (int i = 0; i < 4; ++i) { float d = v[i] - mu; s2 += d * d; }
    red[tid] = s2; __syncthreads();
    for (int off = 128; off > 0; off >>= 1) {
        if (tid < off) red[tid] += red[tid + off];
        __syncthreads();
    }
    float rstd = rsqrtf(red[0] * (1.f / EMB) + 1e-5f);
    #pragma unroll
    for (int i = 0; i < 4; ++i) {
        int c = tid + 256 * i;
        out[(size_t)row * EMB + c] = (_Float16)((v[i] - mu) * rstd * g[c] + b[c]);
    }
}

// ---------------------------------------------------------------------------
// WMMA GEMM: C[M,N] = A[M,K](f16, row-major) * Bt[N,K](f16, n-major)
// Block tile 128x64, 256 threads (8 waves), each wave: 16 rows x 64 cols.
// Double-buffered LDS tiles filled by GLOBAL_LOAD_ASYNC_TO_LDS_B128
// (ASYNCcnt), XOR-swizzled B layout -> conflict-free 2x ds_load_b128 frags.
// mode 0: Cf16 = acc                        (QKV)
// mode 1: Cf32 = acc + bias[n] + Res[m,n]   (proj / FFN2)
// mode 2: Cf16 = relu(acc + bias[n])        (FFN1)
// ---------------------------------------------------------------------------
__global__ void __launch_bounds__(256)
gemm_wmma_kernel(const _Float16* __restrict__ A, const _Float16* __restrict__ Bt,
                 int M, int N, int K, int mode,
                 _Float16* __restrict__ Cf16, float* __restrict__ Cf32,
                 const float* __restrict__ bias, const float* __restrict__ Res) {
#if ASYNC_OK
    __shared__ _Float16 Alds[2][128 * 40];
    __shared__ _Float16 Blds[2][64 * 32];
#else
    __shared__ _Float16 Alds[1][128 * 40];
    __shared__ _Float16 Blds[1][64 * 32];
#endif
    const int tid  = threadIdx.x;
    const int wave = tid >> 5;
    const int lane = tid & 31;
    const int nl   = lane & 15;
    const int hw   = lane >> 4;
    const int mBase = blockIdx.y * 128;
    const int nBase = blockIdx.x * 64;

    v8f acc[4];
    #pragma unroll
    for (int t = 0; t < 4; ++t)
        #pragma unroll
        for (int i = 0; i < 8; ++i) acc[t][i] = 0.f;

    const int nk = K >> 5;

#if ASYNC_OK
    // prologue: async-issue tile 0 (A: 512 16B chunks, B: 256 16B chunks)
    {
        #pragma unroll
        for (int i = 0; i < 2; ++i) {
            int idx = tid + 256 * i;
            int row = idx >> 2, c = idx & 3;
            async_b128(A + (size_t)(mBase + row) * K + c * 8,
                       &Alds[0][row * 40 + c * 8]);
        }
        int row = tid >> 2, c = tid & 3;
        int sw = (row >> 2) & 3;
        async_b128(Bt + (size_t)(nBase + row) * K + c * 8,
                   &Blds[0][row * 32 + ((c ^ sw) << 3)]);
    }
    for (int kt = 0; kt < nk; ++kt) {
        const int cur = kt & 1;
        wait_async0();
        __syncthreads();                 // tile kt resident, all waves past kt-1
        if (kt + 1 < nk) {               // prefetch tile kt+1 into other buffer
            const int nxt = cur ^ 1;
            #pragma unroll
            for (int i = 0; i < 2; ++i) {
                int idx = tid + 256 * i;
                int row = idx >> 2, c = idx & 3;
                async_b128(A + (size_t)(mBase + row) * K + (kt + 1) * 32 + c * 8,
                           &Alds[nxt][row * 40 + c * 8]);
            }
            int row = tid >> 2, c = tid & 3;
            int sw = (row >> 2) & 3;
            async_b128(Bt + (size_t)(nBase + row) * K + (kt + 1) * 32 + c * 8,
                       &Blds[nxt][row * 32 + ((c ^ sw) << 3)]);
        }
        // hoist all fragment loads, then 4 back-to-back WMMAs
        const _Float16* ap = &Alds[cur][(wave * 16 + nl) * 40 + hw * 8];
        v16h a = frag2(ap, ap + 16);
        v16h bb[4];
        #pragma unroll
        for (int t = 0; t < 4; ++t) {
            int n  = t * 16 + nl;
            int sw = (n >> 2) & 3;
            const _Float16* bp = &Blds[cur][n * 32];
            bb[t] = frag2(bp + (((hw)     ^ sw) << 3),
                          bp + (((hw + 2) ^ sw) << 3));
        }
        #pragma unroll
        for (int t = 0; t < 4; ++t)
            acc[t] = __builtin_amdgcn_wmma_f32_16x16x32_f16(
                false, a, false, bb[t], (short)0, acc[t], false, false);
    }
#else
    for (int kt = 0; kt < nk; ++kt) {
        __syncthreads();
        #pragma unroll
        for (int i = 0; i < 8; ++i) {
            int idx = tid + 256 * i;
            int row = idx >> 4;
            int c2  = idx & 15;
            const unsigned int* src =
                (const unsigned int*)(A + (size_t)(mBase + row) * K + kt * 32) + c2;
            *(unsigned int*)(&Alds[0][row * 40 + c2 * 2]) = *src;
        }
        #pragma unroll
        for (int i = 0; i < 4; ++i) {
            int idx = tid + 256 * i;
            int row = idx >> 4;
            int c2  = idx & 15;
            const unsigned int* src =
                (const unsigned int*)(Bt + (size_t)(nBase + row) * K + kt * 32) + c2;
            int blk = c2 >> 2;
            int sw  = (row >> 2) & 3;
            int dwo = ((blk ^ sw) << 2) + (c2 & 3);
            *(unsigned int*)(&Blds[0][row * 32 + dwo * 2]) = *src;
        }
        __syncthreads();
        const _Float16* ap = &Alds[0][(wave * 16 + nl) * 40 + hw * 8];
        v16h a = frag2(ap, ap + 16);
        v16h bb[4];
        #pragma unroll
        for (int t = 0; t < 4; ++t) {
            int n  = t * 16 + nl;
            int sw = (n >> 2) & 3;
            const _Float16* bp = &Blds[0][n * 32];
            bb[t] = frag2(bp + (((hw)     ^ sw) << 3),
                          bp + (((hw + 2) ^ sw) << 3));
        }
        #pragma unroll
        for (int t = 0; t < 4; ++t)
            acc[t] = __builtin_amdgcn_wmma_f32_16x16x32_f16(
                false, a, false, bb[t], (short)0, acc[t], false, false);
    }
#endif

    // epilogue: c[i] -> row = i + 8*hw, col = lane%16
    #pragma unroll
    for (int t = 0; t < 4; ++t) {
        #pragma unroll
        for (int i = 0; i < 8; ++i) {
            int row = mBase + wave * 16 + hw * 8 + i;
            int col = nBase + t * 16 + nl;
            float v = acc[t][i];
            if (mode == 0) {
                Cf16[(size_t)row * N + col] = (_Float16)v;
            } else if (mode == 1) {
                Cf32[(size_t)row * N + col] = v + bias[col] + Res[(size_t)row * N + col];
            } else {
                float u = v + bias[col];
                Cf16[(size_t)row * N + col] = (_Float16)(u > 0.f ? u : 0.f);
            }
        }
    }
}

// ---------------------------------------------------------------------------
// Flash attention (causal). Grid: (SEQ/128, BATCH*HEADS). 256 threads.
// ---------------------------------------------------------------------------
__global__ void __launch_bounds__(256)
attn_kernel(const _Float16* __restrict__ qkv, _Float16* __restrict__ z) {
    __shared__ _Float16 Qlds[128 * 72];   // 144B rows
    __shared__ _Float16 Klds[32 * 72];    // [key][d], contiguous d
    __shared__ _Float16 Vt[64 * 32];      // [d][key], swizzled
    __shared__ _Float16 Plds[8 * 16 * 40];

    const int tid  = threadIdx.x;
    const int wave = tid >> 5;
    const int lane = tid & 31;
    const int nl   = lane & 15;
    const int hw   = lane >> 4;
    const int bh = blockIdx.y;
    const int b  = bh >> 4;
    const int h  = bh & 15;
    const int q0 = blockIdx.x * 128;
    const size_t rs = 3 * EMB;

    // Q tile: 128x64 halves (4096 dwords, 16 per thread)
    #pragma unroll
    for (int i = 0; i < 16; ++i) {
        int idx = tid + 256 * i;
        int r  = idx >> 5;
        int c2 = idx & 31;
        const unsigned int* src =
            (const unsigned int*)(qkv + ((size_t)(b * SEQ + q0 + r)) * rs + h * HD) + c2;
        *(unsigned int*)(Qlds + r * 72 + c2 * 2) = *src;
    }

    float mrow[8], lrow[8];
    v8f o[4];
    #pragma unroll
    for (int i = 0; i < 8; ++i) { mrow[i] = -1e30f; lrow[i] = 0.f; }
    #pragma unroll
    for (int t = 0; t < 4; ++t)
        #pragma unroll
        for (int i = 0; i < 8; ++i) o[t][i] = 0.f;

    const int qrow0 = q0 + wave * 16 + hw * 8;
    const int jmax = (q0 + 127) >> 5;
    for (int j = 0; j <= jmax; ++j) {
        const int t0 = j * 32;
        __syncthreads();
        #pragma unroll
        for (int i = 0; i < 4; ++i) {
            int idx = tid + 256 * i;
            int r  = idx >> 5;         // key
            int c2 = idx & 31;         // dword along d
            size_t base = ((size_t)(b * SEQ + t0 + r)) * rs + h * HD;
            *(unsigned int*)(Klds + r * 72 + c2 * 2) =
                *((const unsigned int*)(qkv + base + EMB) + c2);
            union { unsigned int u; _Float16 hh[2]; } tv;
            tv.u = *((const unsigned int*)(qkv + base + 2 * EMB) + c2);
            int d0 = c2 * 2;
            int blk = r >> 3, pos = r & 7;
            Vt[(d0    ) * 32 + ((blk ^ (((d0    ) >> 2) & 3)) << 3) + pos] = tv.hh[0];
            Vt[(d0 + 1) * 32 + ((blk ^ (((d0 + 1) >> 2) & 3)) << 3) + pos] = tv.hh[1];
        }
        __syncthreads();

        // S = Q K^T : hoist all operands, then 4 back-to-back WMMAs
        const _Float16* qb = Qlds + (wave * 16 + nl) * 72 + hw * 8;
        v16h qa[2];
        qa[0] = frag2(qb,      qb + 16);
        qa[1] = frag2(qb + 32, qb + 48);
        v16h kf[2][2];
        #pragma unroll
        for (int s0 = 0; s0 < 2; ++s0)
            #pragma unroll
            for (int dd = 0; dd < 2; ++dd) {
                const _Float16* kp = Klds + (s0 * 16 + nl) * 72 + dd * 32 + hw * 8;
                kf[s0][dd] = frag2(kp, kp + 16);
            }
        v8f sc[2];
        #pragma unroll
        for (int s0 = 0; s0 < 2; ++s0) {
            #pragma unroll
            for (int i = 0; i < 8; ++i) sc[s0][i] = 0.f;
            #pragma unroll
            for (int dd = 0; dd < 2; ++dd)
                sc[s0] = __builtin_amdgcn_wmma_f32_16x16x32_f16(
                    false, qa[dd], false, kf[s0][dd], (short)0, sc[s0], false, false);
        }

        // scale + causal mask + online softmax
        float rm[8], sval[2][8];
        #pragma unroll
        for (int i = 0; i < 8; ++i) rm[i] = -1e30f;
        #pragma unroll
        for (int s0 = 0; s0 < 2; ++s0) {
            int key = t0 + s0 * 16 + nl;
            #pragma unroll
            for (int i = 0; i < 8; ++i) {
                float s = sc[s0][i] * 0.125f;       // HD^-0.5
                if (key > qrow0 + i) s = -1e30f;
                sval[s0][i] = s;
                rm[i] = fmaxf(rm[i], s);
            }
        }
        #pragma unroll
        for (int m = 1; m < 16; m <<= 1)
            #pragma unroll
            for (int i = 0; i < 8; ++i)
                rm[i] = fmaxf(rm[i], __shfl_xor(rm[i], m, 16));

        float newm[8], scalef[8], rsum[8], pv[2][8];
        #pragma unroll
        for (int i = 0; i < 8; ++i) {
            newm[i]   = fmaxf(mrow[i], rm[i]);
            scalef[i] = __expf(mrow[i] - newm[i]);
            rsum[i]   = 0.f;
        }
        #pragma unroll
        for (int s0 = 0; s0 < 2; ++s0)
            #pragma unroll
            for (int i = 0; i < 8; ++i) {
                float p = __expf(sval[s0][i] - newm[i]);
                pv[s0][i] = p; rsum[i] += p;
            }
        #pragma unroll
        for (int m = 1; m < 16; m <<= 1)
            #pragma unroll
            for (int i = 0; i < 8; ++i)
                rsum[i] += __shfl_xor(rsum[i], m, 16);
        #pragma unroll
        for (int i = 0; i < 8; ++i) {
            lrow[i] = lrow[i] * scalef[i] + rsum[i];
            mrow[i] = newm[i];
        }
        #pragma unroll
        for (int t = 0; t < 4; ++t)
            #pragma unroll
            for (int i = 0; i < 8; ++i) o[t][i] *= scalef[i];

        // P -> per-wave LDS (16x32, stride 40 halves = 80B)
        _Float16* pw = Plds + wave * 16 * 40;
        #pragma unroll
        for (int s0 = 0; s0 < 2; ++s0)
            #pragma unroll
            for (int i = 0; i < 8; ++i)
                pw[(hw * 8 + i) * 40 + s0 * 16 + nl] = (_Float16)pv[s0][i];
        __syncthreads();

        // O += P V : hoist all operands, then 4 back-to-back WMMAs
        const _Float16* pp = pw + nl * 40 + hw * 8;
        v16h a = frag2(pp, pp + 16);
        v16h vf[4];
        #pragma unroll
        for (int t = 0; t < 4; ++t) {
            int n  = t * 16 + nl;          // d index
            int sw = (n >> 2) & 3;
            const _Float16* vp = Vt + n * 32;
            vf[t] = frag2(vp + (((hw)     ^ sw) << 3),
                          vp + (((hw + 2) ^ sw) << 3));
        }
        #pragma unroll
        for (int t = 0; t < 4; ++t)
            o[t] = __builtin_amdgcn_wmma_f32_16x16x32_f16(
                false, a, false, vf[t], (short)0, o[t], false, false);
    }

    // normalize + store z (concat heads layout [ROWS, EMB])
    float inv[8];
    #pragma unroll
    for (int i = 0; i < 8; ++i) inv[i] = 1.f / lrow[i];
    #pragma unroll
    for (int t = 0; t < 4; ++t)
        #pragma unroll
        for (int i = 0; i < 8; ++i) {
            int row = q0 + wave * 16 + hw * 8 + i;
            int col = h * HD + t * 16 + nl;
            z[(size_t)row * EMB + col] = (_Float16)(o[t][i] * inv[i]);
        }
}

// ---------------------------------------------------------------------------
// Launch
// ---------------------------------------------------------------------------
extern "C" void kernel_launch(void* const* d_in, const int* in_sizes, int n_in,
                              void* d_out, int out_size, void* d_ws, size_t ws_size,
                              hipStream_t stream) {
    const float* x      = (const float*)d_in[0];
    const float* wq     = (const float*)d_in[1];
    const float* wk     = (const float*)d_in[2];
    const float* wv     = (const float*)d_in[3];
    const float* w_proj = (const float*)d_in[4];
    const float* b_proj = (const float*)d_in[5];
    const float* w1     = (const float*)d_in[6];
    const float* b1     = (const float*)d_in[7];
    const float* w2     = (const float*)d_in[8];
    const float* b2     = (const float*)d_in[9];
    const float* ln1_g  = (const float*)d_in[10];
    const float* ln1_b  = (const float*)d_in[11];
    const float* ln2_g  = (const float*)d_in[12];
    const float* ln2_b  = (const float*)d_in[13];
    float* out = (float*)d_out;

    char* ws = (char*)d_ws;
    size_t off = 0;
    _Float16* Wqkv = (_Float16*)(ws + off); off += (size_t)EMB * 3 * EMB * 2;       // [3072][1024]
    _Float16* Wp   = (_Float16*)(ws + off); off += (size_t)EMB * EMB * 2;           // [1024][1024]
    _Float16* W1   = (_Float16*)(ws + off); off += (size_t)EMB * FFDIM * 2;         // [2048][1024]
    _Float16* W2   = (_Float16*)(ws + off); off += (size_t)FFDIM * EMB * 2;         // [1024][2048]
    _Float16* h1   = (_Float16*)(ws + off); off += (size_t)ROWS * EMB * 2;
    _Float16* qkvb = (_Float16*)(ws + off); off += (size_t)ROWS * 3 * EMB * 2;
    _Float16* zb   = (_Float16*)(ws + off); off += (size_t)ROWS * EMB * 2;
    float*    x2   = (float*)   (ws + off); off += (size_t)ROWS * EMB * 4;
    _Float16* h2   = (_Float16*)(ws + off); off += (size_t)ROWS * EMB * 2;
    _Float16* ffb  = (_Float16*)(ws + off); off += (size_t)ROWS * FFDIM * 2;

    // 1. weight conversion (transposed to [N][K])
    cvt_qkv_t_kernel<<<(HEADS * EMB * HD) / 256, 256, 0, stream>>>(wq, wk, wv, Wqkv);
    cvt_t_kernel<<<(EMB * EMB) / 256, 256, 0, stream>>>(w_proj, Wp, EMB, EMB);
    cvt_t_kernel<<<(EMB * FFDIM) / 256, 256, 0, stream>>>(w1, W1, FFDIM, EMB);
    cvt_t_kernel<<<(FFDIM * EMB) / 256, 256, 0, stream>>>(w2, W2, EMB, FFDIM);

    // 2. LN1
    layernorm_kernel<<<ROWS, 256, 0, stream>>>(x, ln1_g, ln1_b, h1);

    // 3. QKV GEMM: [8192,1024] x [1024,3072]
    gemm_wmma_kernel<<<dim3(3 * EMB / 64, ROWS / 128), 256, 0, stream>>>(
        h1, Wqkv, ROWS, 3 * EMB, EMB, 0, qkvb, nullptr, nullptr, nullptr);

    // 4. causal flash attention
    attn_kernel<<<dim3(SEQ / 128, BATCH * HEADS), 256, 0, stream>>>(qkvb, zb);

    // 5. output projection + residual: x2 = x + z @ Wp + b_proj
    gemm_wmma_kernel<<<dim3(EMB / 64, ROWS / 128), 256, 0, stream>>>(
        zb, Wp, ROWS, EMB, EMB, 1, nullptr, x2, b_proj, x);

    // 6. LN2
    layernorm_kernel<<<ROWS, 256, 0, stream>>>(x2, ln2_g, ln2_b, h2);

    // 7. FFN1: relu(h2 @ W1 + b1)
    gemm_wmma_kernel<<<dim3(FFDIM / 64, ROWS / 128), 256, 0, stream>>>(
        h2, W1, ROWS, FFDIM, EMB, 2, ffb, nullptr, b1, nullptr);

    // 8. FFN2 + residual -> d_out
    gemm_wmma_kernel<<<dim3(EMB / 64, ROWS / 128), 256, 0, stream>>>(
        ffb, W2, ROWS, EMB, FFDIM, 1, nullptr, out, b2, x2);
}
